// SASRecModel_69380901700358
// MI455X (gfx1250) — matrix operand order repaired and checked by reading, not compiled, not run on hardware
//
#include <hip/hip_runtime.h>
#include <hip/hip_bf16.h>
#include <math.h>

// ---------------------------------------------------------------------------
// Problem constants (from reference): B=64, L=200, H=256, E=8, K(topk)=2
// ---------------------------------------------------------------------------
#define T_TOK   12800      // B*L
#define HD      256        // H
#define HD2     512        // 2H
#define NE      8          // experts
#define LN_EPS  1e-5f

typedef __attribute__((ext_vector_type(16))) __bf16 v16bf;
typedef __attribute__((ext_vector_type(8)))  float  v8f;

union BF16x16 {
    v16bf v;
    unsigned short s[16];
    uint4 q[2];
};

static __device__ __forceinline__ unsigned short f32_to_bf16(float f) {
    union { float f; unsigned u; } v; v.f = f;
    unsigned r = v.u + 0x7FFFu + ((v.u >> 16) & 1u);   // round-to-nearest-even
    return (unsigned short)(r >> 16);
}
static __device__ __forceinline__ float bf16_to_f32(unsigned short s) {
    union { unsigned u; float f; } v; v.u = ((unsigned)s) << 16;
    return v.f;
}

#define WMMA_BF16(acc, a, b) \
    __builtin_amdgcn_wmma_f32_16x16x32_bf16(false, (a), false, (b), (short)0, (acc), false, false)

// ---------------------------------------------------------------------------
// Elementwise f32 -> bf16 (activations). 8 elements / thread, vector ld/st.
// ---------------------------------------------------------------------------
__global__ __launch_bounds__(256)
void cvt_bf16_kernel(const float* __restrict__ src,
                     unsigned short* __restrict__ dst, int total8) {
    int i = blockIdx.x * blockDim.x + threadIdx.x;
    if (i >= total8) return;
    const float* p = src + i * 8;
    float4 f0 = *(const float4*)(p);
    float4 f1 = *(const float4*)(p + 4);
    union { uint4 q; unsigned short s[8]; } o;
    o.s[0] = f32_to_bf16(f0.x); o.s[1] = f32_to_bf16(f0.y);
    o.s[2] = f32_to_bf16(f0.z); o.s[3] = f32_to_bf16(f0.w);
    o.s[4] = f32_to_bf16(f1.x); o.s[5] = f32_to_bf16(f1.y);
    o.s[6] = f32_to_bf16(f1.z); o.s[7] = f32_to_bf16(f1.w);
    *(uint4*)(dst + i * 8) = o.q;
}

// ---------------------------------------------------------------------------
// Weight prep: f32 [batch][K][N] -> bf16 transposed [batch][N][K]
// (so WMMA B-fragments are contiguous per-lane 32B loads)
// ---------------------------------------------------------------------------
__global__ void cvt_transpose_kernel(const float* __restrict__ src,
                                     unsigned short* __restrict__ dst,
                                     int K, int N, int total) {
    int idx = blockIdx.x * blockDim.x + threadIdx.x;
    if (idx >= total) return;
    int kn = K * N;
    int b  = idx / kn;
    int r  = idx - b * kn;
    int k  = r / N;
    int n  = r - k * N;
    dst[b * kn + n * K + k] = f32_to_bf16(src[idx]);
}

// ---------------------------------------------------------------------------
// Fragment loaders (A: row-major [rows][K] bf16; B: transposed [N][K] bf16)
// A per-lane: 16 contiguous bf16 @ row*K + k0 + sel*16     (two b128 loads)
// B per-lane: 16 contiguous bf16 @ col*K + k0 + sel*16     (two b128 loads)
// ---------------------------------------------------------------------------
static __device__ __forceinline__ BF16x16 load_a(const unsigned short* base) {
    BF16x16 a;
    a.q[0] = *(const uint4*)(base);
    a.q[1] = *(const uint4*)(base + 8);
    return a;
}
static __device__ __forceinline__ BF16x16 load_b(const unsigned short* base) {
    BF16x16 b;
    b.q[0] = *(const uint4*)(base);
    b.q[1] = *(const uint4*)(base + 8);
    return b;
}

// ---------------------------------------------------------------------------
// Stage 1: mu = emb@W_mu + b_mu ; sigma = exp(emb@W_sg + b_sg) ;
//          z = mu + sigma*noise  (z stored as bf16 for later WMMA stages)
// One 16x16 output tile per wave; A fragment shared by both WMMA chains.
// ---------------------------------------------------------------------------
__global__ __launch_bounds__(256)
void stage1_reparam_kernel(const unsigned short* __restrict__ emb,     // [T][H] bf16
                           const unsigned short* __restrict__ WTmu,    // [H][H] bf16 (n-major)
                           const unsigned short* __restrict__ WTsg,
                           const float* __restrict__ bmu,
                           const float* __restrict__ bsg,
                           const float* __restrict__ noise,
                           float* __restrict__ out_mu,
                           float* __restrict__ out_sg,
                           unsigned short* __restrict__ z_bf16) {
    const int wave = (blockIdx.x * blockDim.x + threadIdx.x) >> 5;   // 12800 tiles
    const int lane = threadIdx.x & 31;
    const int sel  = lane >> 4;        // half-wave select
    const int lm   = lane & 15;
    const int tm   = wave >> 4;        // token tile (800)
    const int tn   = wave & 15;        // n tile (16)
    const int tok0 = tm * 16;
    const int n0   = tn * 16;
    const int rowA = tok0 + lm;
    const int colB = n0 + lm;

    v8f acc_mu = {0.f,0.f,0.f,0.f,0.f,0.f,0.f,0.f};
    v8f acc_sg = {0.f,0.f,0.f,0.f,0.f,0.f,0.f,0.f};

    for (int kt = 0; kt < 8; ++kt) {
        const int k0 = kt * 32;
        BF16x16 a  = load_a(emb  + rowA * HD + k0 + sel * 16);
        BF16x16 b0 = load_b(WTmu + colB * HD + k0 + sel * 16);
        BF16x16 b1 = load_b(WTsg + colB * HD + k0 + sel * 16);
        acc_mu = WMMA_BF16(acc_mu, a.v, b0.v);
        acc_sg = WMMA_BF16(acc_sg, a.v, b1.v);
    }

    const float bm = bmu[colB];
    const float bs = bsg[colB];
    #pragma unroll
    for (int i = 0; i < 8; ++i) {
        const int row = tok0 + i + sel * 8;     // C layout: row = VGPR idx + 8*half
        const int idx = row * HD + colB;
        float mu = acc_mu[i] + bm;
        float sg = expf(acc_sg[i] + bs);
        float zv = mu + sg * noise[idx];
        out_mu[idx] = mu;
        out_sg[idx] = sg;
        z_bf16[idx] = f32_to_bf16(zv);
    }
}

// ---------------------------------------------------------------------------
// Gating: softmax(z@W_gate + b_gate) -> top-2 -> renormalize. One thread/token.
// ---------------------------------------------------------------------------
__global__ __launch_bounds__(256)
void gate_kernel(const unsigned short* __restrict__ z,
                 const float* __restrict__ Wg,       // [H][E]
                 const float* __restrict__ bg,       // [E]
                 float* __restrict__ gate) {         // [T][E]
    int t = blockIdx.x * blockDim.x + threadIdx.x;
    if (t >= T_TOK) return;
    float s[NE];
    #pragma unroll
    for (int e = 0; e < NE; ++e) s[e] = bg[e];
    const unsigned short* zr = z + t * HD;
    for (int h = 0; h < HD; ++h) {
        float zh = bf16_to_f32(zr[h]);
        const float* w = Wg + h * NE;
        #pragma unroll
        for (int e = 0; e < NE; ++e) s[e] += zh * w[e];
    }
    float m = s[0];
    #pragma unroll
    for (int e = 1; e < NE; ++e) m = fmaxf(m, s[e]);
    float w[NE], wsum = 0.f;
    #pragma unroll
    for (int e = 0; e < NE; ++e) { w[e] = expf(s[e] - m); wsum += w[e]; }
    #pragma unroll
    for (int e = 0; e < NE; ++e) w[e] /= wsum;
    int i1 = 0;
    #pragma unroll
    for (int e = 1; e < NE; ++e) if (w[e] > w[i1]) i1 = e;
    int i2 = (i1 == 0) ? 1 : 0;
    #pragma unroll
    for (int e = 0; e < NE; ++e) if (e != i1 && w[e] > w[i2]) i2 = e;
    float den = w[i1] + w[i2] + 1e-8f;
    #pragma unroll
    for (int e = 0; e < NE; ++e)
        gate[t * NE + e] = (e == i1 || e == i2) ? (w[e] / den) : 0.f;
}

// ---------------------------------------------------------------------------
// Expert stack + gated combine: out = sum_e gate_e * (z @ W_e + b_e)
// k-outer / expert-inner: A fragment loaded once per k-step, reused by 8
// independent WMMA chains (8 resident accumulators).
// ---------------------------------------------------------------------------
__global__ __launch_bounds__(256)
void expert_combine_kernel(const unsigned short* __restrict__ z,     // [T][H] bf16
                           const unsigned short* __restrict__ WT,    // [E][H][H] bf16 (n-major)
                           const float* __restrict__ bexp,           // [E][H]
                           const float* __restrict__ gate,           // [T][E]
                           unsigned short* __restrict__ cat,         // [T][2H] bf16
                           int colofs) {
    const int wave = (blockIdx.x * blockDim.x + threadIdx.x) >> 5;
    const int lane = threadIdx.x & 31;
    const int sel  = lane >> 4;
    const int lm   = lane & 15;
    const int tm   = wave >> 4;
    const int tn   = wave & 15;
    const int tok0 = tm * 16;
    const int n0   = tn * 16;
    const int rowA = tok0 + lm;
    const int colB = n0 + lm;

    v8f acc[NE];
    #pragma unroll
    for (int e = 0; e < NE; ++e) acc[e] = (v8f){0.f,0.f,0.f,0.f,0.f,0.f,0.f,0.f};

    for (int kt = 0; kt < 8; ++kt) {
        const int k0 = kt * 32;
        BF16x16 a = load_a(z + rowA * HD + k0 + sel * 16);
        #pragma unroll
        for (int e = 0; e < NE; ++e) {
            BF16x16 b = load_b(WT + e * HD * HD + colB * HD + k0 + sel * 16);
            acc[e] = WMMA_BF16(acc[e], a.v, b.v);
        }
    }

    float tot[8] = {0.f,0.f,0.f,0.f,0.f,0.f,0.f,0.f};
    #pragma unroll
    for (int e = 0; e < NE; ++e) {
        const float bias = bexp[e * HD + colB];
        #pragma unroll
        for (int i = 0; i < 8; ++i) {
            float g = gate[(tok0 + i + sel * 8) * NE + e];
            tot[i] += g * (acc[e][i] + bias);
        }
    }
    #pragma unroll
    for (int i = 0; i < 8; ++i) {
        const int row = tok0 + i + sel * 8;
        cat[row * HD2 + colofs + colB] = f32_to_bf16(tot[i]);
    }
}

// ---------------------------------------------------------------------------
// Fusion: cat @ W_fuse + b_fuse -> LayerNorm -> ReLU -> + item_emb
// WG handles 16 full token rows (all 256 cols) so LN can see whole rows.
// 8 waves x 2 n-tiles each; results staged in LDS; wave-parallel LN after.
// ---------------------------------------------------------------------------
__global__ __launch_bounds__(256)
void fuse_ln_kernel(const unsigned short* __restrict__ cat,   // [T][2H] bf16
                    const unsigned short* __restrict__ WTf,   // [H][2H] bf16 (n-major)
                    const float* __restrict__ bfu,            // [H]
                    const float* __restrict__ ln_g,
                    const float* __restrict__ ln_b,
                    const float* __restrict__ item,
                    float* __restrict__ out) {
    __shared__ float tile[16 * HD];     // 16 KB
    const int tok0 = blockIdx.x * 16;
    const int wave = threadIdx.x >> 5;
    const int lane = threadIdx.x & 31;
    const int sel  = lane >> 4;
    const int lm   = lane & 15;
    const int rowA = tok0 + lm;

    #pragma unroll
    for (int t = 0; t < 2; ++t) {
        const int n0 = (wave * 2 + t) * 16;
        const int colB = n0 + lm;
        v8f acc = {0.f,0.f,0.f,0.f,0.f,0.f,0.f,0.f};
        for (int kt = 0; kt < 16; ++kt) {     // K = 512
            const int k0 = kt * 32;
            BF16x16 a = load_a(cat + rowA * HD2 + k0 + sel * 16);
            BF16x16 b = load_b(WTf + colB * HD2 + k0 + sel * 16);
            acc = WMMA_BF16(acc, a.v, b.v);
        }
        const float bb = bfu[colB];
        #pragma unroll
        for (int i = 0; i < 8; ++i)
            tile[(i + sel * 8) * HD + colB] = acc[i] + bb;
    }
    __syncthreads();

    // LayerNorm + ReLU + residual: wave w handles rows 2w, 2w+1 (wave32 shfl)
    #pragma unroll
    for (int t = 0; t < 2; ++t) {
        const int r = wave * 2 + t;
        const float* rowp = tile + r * HD;
        float x[8], s = 0.f;
        #pragma unroll
        for (int c = 0; c < 8; ++c) { x[c] = rowp[lane * 8 + c]; s += x[c]; }
        #pragma unroll
        for (int m = 16; m >= 1; m >>= 1) s += __shfl_xor(s, m, 32);
        const float mu = s * (1.f / 256.f);
        float v = 0.f;
        #pragma unroll
        for (int c = 0; c < 8; ++c) { float d = x[c] - mu; v += d * d; }
        #pragma unroll
        for (int m = 16; m >= 1; m >>= 1) v += __shfl_xor(v, m, 32);
        const float rstd = rsqrtf(v * (1.f / 256.f) + LN_EPS);
        const int row = tok0 + r;
        #pragma unroll
        for (int c = 0; c < 8; ++c) {
            const int col = lane * 8 + c;
            float y = (x[c] - mu) * rstd * ln_g[col] + ln_b[col];
            y = fmaxf(y, 0.f);
            out[row * HD + col] = item[row * HD + col] + y;
        }
    }
}

// ---------------------------------------------------------------------------
// Host-side launch
// ---------------------------------------------------------------------------
extern "C" void kernel_launch(void* const* d_in, const int* in_sizes, int n_in,
                              void* d_out, int out_size, void* d_ws, size_t ws_size,
                              hipStream_t stream) {
    const float* text_emb = (const float*)d_in[0];
    const float* img_emb  = (const float*)d_in[1];
    const float* item_emb = (const float*)d_in[2];
    const float* noise_t  = (const float*)d_in[3];
    const float* noise_i  = (const float*)d_in[4];
    const float* W_mu_t = (const float*)d_in[5];  const float* b_mu_t = (const float*)d_in[6];
    const float* W_sg_t = (const float*)d_in[7];  const float* b_sg_t = (const float*)d_in[8];
    const float* W_mu_i = (const float*)d_in[9];  const float* b_mu_i = (const float*)d_in[10];
    const float* W_sg_i = (const float*)d_in[11]; const float* b_sg_i = (const float*)d_in[12];
    const float* W_gate = (const float*)d_in[13]; const float* b_gate = (const float*)d_in[14];
    const float* W_texp = (const float*)d_in[15]; const float* b_texp = (const float*)d_in[16];
    const float* W_iexp = (const float*)d_in[17]; const float* b_iexp = (const float*)d_in[18];
    const float* W_fuse = (const float*)d_in[19]; const float* b_fuse = (const float*)d_in[20];
    const float* ln_g   = (const float*)d_in[21]; const float* ln_b   = (const float*)d_in[22];

    float* out      = (float*)d_out;
    float* fusion_o = out;
    float* t_mu_o   = out + 1 * T_TOK * HD;
    float* t_sg_o   = out + 2 * T_TOK * HD;
    float* i_mu_o   = out + 3 * T_TOK * HD;
    float* i_sg_o   = out + 4 * T_TOK * HD;

    // ---- workspace carve-up (bf16 stored as ushort) ----
    char* ws = (char*)d_ws;
    unsigned short* text_bf = (unsigned short*)ws; ws += (size_t)T_TOK * HD * 2;
    unsigned short* img_bf  = (unsigned short*)ws; ws += (size_t)T_TOK * HD * 2;
    unsigned short* zt  = (unsigned short*)ws; ws += (size_t)T_TOK * HD * 2;
    unsigned short* zi  = (unsigned short*)ws; ws += (size_t)T_TOK * HD * 2;
    unsigned short* cat = (unsigned short*)ws; ws += (size_t)T_TOK * HD2 * 2;
    unsigned short* WTmu_t = (unsigned short*)ws; ws += (size_t)HD * HD * 2;
    unsigned short* WTsg_t = (unsigned short*)ws; ws += (size_t)HD * HD * 2;
    unsigned short* WTmu_i = (unsigned short*)ws; ws += (size_t)HD * HD * 2;
    unsigned short* WTsg_i = (unsigned short*)ws; ws += (size_t)HD * HD * 2;
    unsigned short* WTtexp = (unsigned short*)ws; ws += (size_t)NE * HD * HD * 2;
    unsigned short* WTiexp = (unsigned short*)ws; ws += (size_t)NE * HD * HD * 2;
    unsigned short* WTfuse = (unsigned short*)ws; ws += (size_t)HD2 * HD * 2;
    float* gate_t = (float*)ws; ws += (size_t)T_TOK * NE * 4;
    float* gate_i = (float*)ws; ws += (size_t)T_TOK * NE * 4;

    // ---- 1) prep: activations f32->bf16, weights transpose+cvt ----
    {
        int t8 = T_TOK * HD / 8;
        cvt_bf16_kernel<<<(t8 + 255) / 256, 256, 0, stream>>>(text_emb, text_bf, t8);
        cvt_bf16_kernel<<<(t8 + 255) / 256, 256, 0, stream>>>(img_emb,  img_bf,  t8);
        int n1 = HD * HD;
        cvt_transpose_kernel<<<(n1 + 255) / 256, 256, 0, stream>>>(W_mu_t, WTmu_t, HD, HD, n1);
        cvt_transpose_kernel<<<(n1 + 255) / 256, 256, 0, stream>>>(W_sg_t, WTsg_t, HD, HD, n1);
        cvt_transpose_kernel<<<(n1 + 255) / 256, 256, 0, stream>>>(W_mu_i, WTmu_i, HD, HD, n1);
        cvt_transpose_kernel<<<(n1 + 255) / 256, 256, 0, stream>>>(W_sg_i, WTsg_i, HD, HD, n1);
        int ne = NE * HD * HD;
        cvt_transpose_kernel<<<(ne + 255) / 256, 256, 0, stream>>>(W_texp, WTtexp, HD, HD, ne);
        cvt_transpose_kernel<<<(ne + 255) / 256, 256, 0, stream>>>(W_iexp, WTiexp, HD, HD, ne);
        int nf = HD2 * HD;
        cvt_transpose_kernel<<<(nf + 255) / 256, 256, 0, stream>>>(W_fuse, WTfuse, HD2, HD, nf);
    }

    // ---- 2) reparameterization GEMMs (WMMA bf16) ----
    // tiles = (T/16)*(H/16) = 12800 waves -> 1600 blocks of 8 waves
    stage1_reparam_kernel<<<1600, 256, 0, stream>>>(text_bf, WTmu_t, WTsg_t, b_mu_t, b_sg_t,
                                                    noise_t, t_mu_o, t_sg_o, zt);
    stage1_reparam_kernel<<<1600, 256, 0, stream>>>(img_bf, WTmu_i, WTsg_i, b_mu_i, b_sg_i,
                                                    noise_i, i_mu_o, i_sg_o, zi);

    // ---- 3) top-2 gating ----
    gate_kernel<<<(T_TOK + 255) / 256, 256, 0, stream>>>(zt, W_gate, b_gate, gate_t);
    gate_kernel<<<(T_TOK + 255) / 256, 256, 0, stream>>>(zi, W_gate, b_gate, gate_i);

    // ---- 4) expert stacks + gated combine into concatenated buffer ----
    expert_combine_kernel<<<1600, 256, 0, stream>>>(zt, WTtexp, b_texp, gate_t, cat, 0);
    expert_combine_kernel<<<1600, 256, 0, stream>>>(zi, WTiexp, b_iexp, gate_i, cat, HD);

    // ---- 5) fusion GEMM + LayerNorm + ReLU + residual ----
    fuse_ln_kernel<<<T_TOK / 16, 256, 0, stream>>>(cat, WTfuse, b_fuse, ln_g, ln_b,
                                                   item_emb, fusion_o);
}
